// InterestDictSoft_euc2_71511205478467
// MI455X (gfx1250) — compile-verified, matrix operand order checked by reading, not run on hardware
//
#include <hip/hip_runtime.h>
#include <hip/hip_bf16.h>
#include <math.h>

typedef __attribute__((ext_vector_type(16))) __bf16 v16bf;
typedef __attribute__((ext_vector_type(8)))  __bf16 v8bf;
typedef __attribute__((ext_vector_type(8)))  float  v8f;
typedef __attribute__((ext_vector_type(4)))  float  v4f;

#define C_DICT   512
#define D_DIM    128
#define TOPK     8
#define WAVES    4
#define ITERS    4
#define ROWS_PER_BLOCK (WAVES * 16 * ITERS)   // 256
#define COS_EPS  1e-8f
#define DECAY    0.1f
#define SM_SCALE 20.0f
#define NB2      128        // blocks in scatter-GEMM pass
#define NCHUNK   32         // rows (K) per scatter-GEMM chunk

// ---------------------------------------------------------------------------
// Kernel 1: normalize dictionary rows -> bf16 d_hat in ws; zero accumulators.
// ---------------------------------------------------------------------------
__global__ void __launch_bounds__(128)
prep_kernel(const float* __restrict__ dict,
            __bf16* __restrict__ dhat,
            float* __restrict__ cluster,
            float* __restrict__ input_sum) {
  const int c = blockIdx.x;
  const int t = threadIdx.x;            // 0..127
  __shared__ float partial[WAVES];
  __shared__ float inv_s;

  float v = dict[c * D_DIM + t];
  float s = v * v;
  #pragma unroll
  for (int off = 16; off > 0; off >>= 1) s += __shfl_down(s, off);
  if ((t & 31) == 0) partial[t >> 5] = s;
  __syncthreads();
  if (t == 0) {
    float tot = partial[0] + partial[1] + partial[2] + partial[3];
    inv_s = 1.0f / fmaxf(sqrtf(tot), COS_EPS);
    cluster[c] = 0.0f;
  }
  __syncthreads();
  dhat[c * D_DIM + t] = (__bf16)(v * inv_s);
  input_sum[c * D_DIM + t] = 0.0f;
}

// ---------------------------------------------------------------------------
// Kernel 2: fused cosine-GEMM (bf16 WMMA) + top-8 + softmaxes + group_emb.
// Packs (idx, bf16 encoding-weight) per row for the pass-3 scatter-GEMM.
// block = 128 threads (4 waves); each wave handles 16 rows x 4 iterations.
// ---------------------------------------------------------------------------
__global__ void __launch_bounds__(WAVES * 32)
main_kernel(const float* __restrict__ x,
            const float* __restrict__ dict,
            const __bf16* __restrict__ dhat,
            float* __restrict__ out_ge,
            float* __restrict__ out_ti,
            float* __restrict__ out_dist,
            float* __restrict__ cluster,
            unsigned int* __restrict__ topk_pack,
            int N) {
  __shared__ __bf16 xa[WAVES][16][136];        // normalized x tile (padded)
  __shared__ float  dist_s[WAVES][16][516];    // distance strip (padded)
  __shared__ float  topw_s[WAVES][16][TOPK];
  __shared__ int    topi_s[WAVES][16][TOPK];
  __shared__ float  clus_s[C_DICT];            // per-block cluster_size partial

  const int tid  = threadIdx.x;
  const int w    = tid >> 5;
  const int lane = tid & 31;
  const int r    = lane & 15;
  const int hi   = lane >> 4;                  // 0 = lanes 0-15, 1 = lanes 16-31

  for (int i = tid; i < C_DICT; i += WAVES * 32) clus_s[i] = 0.0f;
  __syncthreads();

  for (int it = 0; it < ITERS; ++it) {
    const int rowBase = blockIdx.x * ROWS_PER_BLOCK + it * (WAVES * 16) + w * 16;
    const bool valid  = (rowBase + 16) <= N;

    // ---- Phase 1: load 16 rows of x, L2-normalize, stash bf16 in LDS ----
    if (valid) {
      const int xr = lane >> 1;                // row within tile: 0..15
      const int xh = lane & 1;                 // which 64-column half
      const float* xp = x + (size_t)(rowBase + xr) * D_DIM + xh * 64;
      float sumsq = 0.f;
      #pragma unroll
      for (int j = 0; j < 16; ++j) {
        float4 b = ((const float4*)xp)[j];
        sumsq += b.x * b.x + b.y * b.y + b.z * b.z + b.w * b.w;
      }
      sumsq += __shfl_xor(sumsq, 1);           // combine halves of the row
      const float inv = 1.0f / fmaxf(sqrtf(sumsq), COS_EPS);
      #pragma unroll
      for (int j = 0; j < 16; ++j) {
        float4 b = ((const float4*)xp)[j];     // L1 hit
        const int cb = xh * 64 + j * 4;
        xa[w][xr][cb + 0] = (__bf16)(b.x * inv);
        xa[w][xr][cb + 1] = (__bf16)(b.y * inv);
        xa[w][xr][cb + 2] = (__bf16)(b.z * inv);
        xa[w][xr][cb + 3] = (__bf16)(b.w * inv);
      }
    }
    __syncthreads();

    // ---- Phase 2: WMMA 16x512 cosine strip ----
    if (valid) {
      v16bf a[4];
      #pragma unroll
      for (int kk = 0; kk < 4; ++kk) {
        const __bf16* ap = &xa[w][r][0];
        const int k0 = kk * 32 + hi * 8;
        const int k1 = kk * 32 + 16 + hi * 8;
        v8bf lo = *(const v8bf*)(ap + k0);
        v8bf hh = *(const v8bf*)(ap + k1);
        a[kk] = __builtin_shufflevector(lo, hh,
                 0,1,2,3,4,5,6,7,8,9,10,11,12,13,14,15);
      }
      for (int ct = 0; ct < 32; ++ct) {
        v8f acc = {};
        const __bf16* bp = dhat + (size_t)(ct * 16 + r) * D_DIM;
        #pragma unroll
        for (int kk = 0; kk < 4; ++kk) {
          const int k0 = kk * 32 + hi * 8;
          const int k1 = kk * 32 + 16 + hi * 8;
          v8bf lo = *(const v8bf*)(bp + k0);
          v8bf hh = *(const v8bf*)(bp + k1);
          v16bf b = __builtin_shufflevector(lo, hh,
                     0,1,2,3,4,5,6,7,8,9,10,11,12,13,14,15);
          acc = __builtin_amdgcn_wmma_f32_16x16x32_bf16(
                  false, a[kk], false, b, (short)0, acc, false, false);
        }
        const int col = ct * 16 + r;
        #pragma unroll
        for (int i = 0; i < 8; ++i) {
          const int rr = hi * 8 + i;
          dist_s[w][rr][col] = acc[i];
          // streamed output, never re-read on device -> non-temporal store
          __builtin_nontemporal_store(
              acc[i], &out_dist[(size_t)(rowBase + rr) * C_DICT + col]);
        }
      }
    }
    __syncthreads();

    // ---- Phase 3: top-8 per row (float4 scan) + softmaxes + packing ----
    if (valid) {
      float tv[TOPK]; int tix[TOPK];
      #pragma unroll
      for (int j = 0; j < TOPK; ++j) { tv[j] = -1e30f; tix[j] = 0; }
      const int cb = hi * 256;                 // 2 lanes per row

#define TOPK_INSERT(VAL, IDX)                                             \
      do {                                                                \
        if ((VAL) > tv[TOPK - 1]) {                                       \
          tv[TOPK - 1] = (VAL); tix[TOPK - 1] = (IDX);                    \
          _Pragma("unroll")                                               \
          for (int _j = TOPK - 1; _j > 0; --_j)                           \
            if (tv[_j] > tv[_j - 1]) {                                    \
              float _tf = tv[_j]; tv[_j] = tv[_j-1]; tv[_j-1] = _tf;      \
              int   _tt = tix[_j]; tix[_j] = tix[_j-1]; tix[_j-1] = _tt;  \
            }                                                             \
        }                                                                 \
      } while (0)

      const v4f* dp = (const v4f*)&dist_s[w][r][cb];
      for (int c4 = 0; c4 < 64; ++c4) {
        const v4f f = dp[c4];
        const float m4 = fmaxf(fmaxf(f.x, f.y), fmaxf(f.z, f.w));
        if (m4 > tv[TOPK - 1]) {               // rare with random data
          const int b4 = cb + c4 * 4;
          TOPK_INSERT(f.x, b4 + 0);
          TOPK_INSERT(f.y, b4 + 1);
          TOPK_INSERT(f.z, b4 + 2);
          TOPK_INSERT(f.w, b4 + 3);
        }
      }
      // merge the two half-scans into lanes 0..15
      #pragma unroll
      for (int j = 0; j < TOPK; ++j) {
        const float ov = __shfl(tv[j],  lane + 16);
        const int   oi = __shfl(tix[j], lane + 16);
        if (hi == 0) TOPK_INSERT(ov, oi);
      }
#undef TOPK_INSERT
      if (hi == 0) {
        const float m = tv[0];
        float ew[TOPK], ee[TOPK], sw = 0.f, se = 0.f;
        #pragma unroll
        for (int j = 0; j < TOPK; ++j) {
          ew[j] = __expf(tv[j] - m);              sw += ew[j];
          ee[j] = __expf(SM_SCALE * (tv[j] - m)); se += ee[j];
        }
        const float isw = 1.0f / sw, ise = 1.0f / se;
        #pragma unroll
        for (int j = 0; j < TOPK; ++j) {
          const float wj = ew[j] * isw;
          const float ej = ee[j] * ise;
          topw_s[w][r][j] = wj;
          topi_s[w][r][j] = tix[j];
          out_ti[(size_t)(rowBase + r) * TOPK + j] = (float)tix[j];
          __bf16 eh = (__bf16)ej;
          const unsigned int pk = ((unsigned int)tix[j] << 16) |
                                  (unsigned int)__builtin_bit_cast(unsigned short, eh);
          topk_pack[(size_t)(rowBase + r) * TOPK + j] = pk;
          atomicAdd(&clus_s[tix[j]], ej);        // ds_add_f32, per-block partial
        }
      }
    }
    __syncthreads();

    // ---- Phase 4: group_emb gather (2 lanes / row) ----
    if (valid) {
      float wv[TOPK]; int iv[TOPK];
      #pragma unroll
      for (int j = 0; j < TOPK; ++j) {
        wv[j] = topw_s[w][r][j];
        iv[j] = topi_s[w][r][j];
      }
      float* gout = out_ge + (size_t)(rowBase + r) * D_DIM + hi * 64;
      #pragma unroll 4
      for (int j4 = 0; j4 < 16; ++j4) {
        const int col = hi * 64 + j4 * 4;
        v4f acc = {0.f, 0.f, 0.f, 0.f};
        #pragma unroll
        for (int k = 0; k < TOPK; ++k) {
          const v4f dv = *(const v4f*)(dict + (size_t)iv[k] * D_DIM + col);
          acc.x += wv[k] * dv.x; acc.y += wv[k] * dv.y;
          acc.z += wv[k] * dv.z; acc.w += wv[k] * dv.w;
        }
        __builtin_nontemporal_store(acc, (v4f*)gout + j4);
      }
    }
    __syncthreads();
  }

  // flush per-block cluster_size partial
  for (int i = tid; i < C_DICT; i += WAVES * 32) {
    const float v = clus_s[i];
    if (v != 0.0f) atomicAdd(&cluster[i], v);
  }
}

// ---------------------------------------------------------------------------
// Kernel 3: input_sum = E^T @ x as a bf16 WMMA GEMM over K = rows.
// 128 blocks x 8 waves; each wave owns a 64x128 slab of the [512,128] output
// kept in 256 accumulator VGPRs; one atomic flush per block at the end.
// ---------------------------------------------------------------------------
__global__ void __launch_bounds__(256)
scatter_gemm_kernel(const float* __restrict__ x,
                    const unsigned int* __restrict__ topk_pack,
                    float* __restrict__ input_sum,
                    int N) {
  __shared__ unsigned short eT[C_DICT][40];   // E^T tile, bf16 bits (padded)
  __shared__ unsigned short xbT[D_DIM][40];   // x^T tile, bf16 bits (padded)

  const int tid  = threadIdx.x;               // 0..255
  const int w    = tid >> 5;                  // 8 waves
  const int lane = tid & 31;
  const int r    = lane & 15;
  const int hi   = lane >> 4;

  for (int i = tid; i < C_DICT * 40; i += 256) ((unsigned short*)eT)[i] = 0;
  __syncthreads();

  v8f acc[4][8];
  #pragma unroll
  for (int ct = 0; ct < 4; ++ct)
    #pragma unroll
    for (int dt = 0; dt < 8; ++dt) acc[ct][dt] = (v8f){};

  const int rowsPer = N / NB2;                // 2048 (N % (NB2*NCHUNK) == 0)
  const int nStart  = blockIdx.x * rowsPer;
  const int nEnd    = nStart + rowsPer;

  for (int nb = nStart; nb + NCHUNK <= nEnd; nb += NCHUNK) {
    // scatter 256 packed (idx, e) entries: thread t -> (row nb + t/8, k = t%8)
    const unsigned int pk = topk_pack[(size_t)nb * TOPK + tid];
    const int eRow = (int)(pk >> 16);
    const int eN   = tid >> 3;
    eT[eRow][eN] = (unsigned short)(pk & 0xffffu);

    // stage x^T bf16 (coalesced global reads, transposed LDS writes)
    const int nl0 = tid >> 7;                 // 0..1
    const int dd  = tid & 127;
    #pragma unroll
    for (int j = 0; j < NCHUNK / 2; ++j) {
      const int nl = nl0 + j * 2;
      __bf16 hv = (__bf16)x[(size_t)(nb + nl) * D_DIM + dd];
      xbT[dd][nl] = __builtin_bit_cast(unsigned short, hv);
    }
    __syncthreads();

    // B fragments (x^T), reused by all 4 of this wave's c-tiles
    v16bf bfr[8];
    #pragma unroll
    for (int dt = 0; dt < 8; ++dt) {
      const unsigned short* bp = &xbT[dt * 16 + r][0];
      v8bf lo = *(const v8bf*)(bp + hi * 8);
      v8bf hh = *(const v8bf*)(bp + 16 + hi * 8);
      bfr[dt] = __builtin_shufflevector(lo, hh,
                 0,1,2,3,4,5,6,7,8,9,10,11,12,13,14,15);
    }
    #pragma unroll
    for (int ct = 0; ct < 4; ++ct) {
      const unsigned short* ap = &eT[w * 64 + ct * 16 + r][0];
      v8bf lo = *(const v8bf*)(ap + hi * 8);
      v8bf hh = *(const v8bf*)(ap + 16 + hi * 8);
      v16bf afr = __builtin_shufflevector(lo, hh,
                   0,1,2,3,4,5,6,7,8,9,10,11,12,13,14,15);
      #pragma unroll
      for (int dt = 0; dt < 8; ++dt)
        acc[ct][dt] = __builtin_amdgcn_wmma_f32_16x16x32_bf16(
            false, afr, false, bfr[dt], (short)0, acc[ct][dt], false, false);
    }
    __syncthreads();

    eT[eRow][eN] = 0;                         // clear only what we scattered
    __syncthreads();
  }

  // flush this wave's 64x128 slab (256 atomics per lane, 65536 per block)
  #pragma unroll
  for (int ct = 0; ct < 4; ++ct)
    #pragma unroll
    for (int dt = 0; dt < 8; ++dt)
      #pragma unroll
      for (int i = 0; i < 8; ++i)
        atomicAdd(&input_sum[(size_t)(w * 64 + ct * 16 + hi * 8 + i) * D_DIM +
                             dt * 16 + r],
                  acc[ct][dt][i]);
}

// ---------------------------------------------------------------------------
// Kernel 4: masked EMA dictionary update.
// ---------------------------------------------------------------------------
__global__ void __launch_bounds__(128)
finalize_kernel(const float* __restrict__ dict,
                const float* __restrict__ cluster,
                const float* __restrict__ input_sum,
                float* __restrict__ out_nd) {
  const int c = blockIdx.x;
  const int t = threadIdx.x;
  const float size = cluster[c];
  const float d    = dict[c * D_DIM + t];
  const float s    = input_sum[c * D_DIM + t];
  float nd = d;
  if (size != 0.0f) nd = d * DECAY + (s / size) * (1.0f - DECAY);
  out_nd[c * D_DIM + t] = nd;
}

// ---------------------------------------------------------------------------
extern "C" void kernel_launch(void* const* d_in, const int* in_sizes, int n_in,
                              void* d_out, int out_size, void* d_ws, size_t ws_size,
                              hipStream_t stream) {
  const float* x    = (const float*)d_in[0];   // [N, 128]
  const float* dict = (const float*)d_in[1];   // [512, 128]
  const int N = in_sizes[0] / D_DIM;

  // workspace layout
  char*         ws        = (char*)d_ws;
  __bf16*       dhat      = (__bf16*)ws;                    // 131072 B
  float*        cluster   = (float*)(ws + 131072);          //   2048 B
  float*        input_sum = (float*)(ws + 133120);          // 262144 B
  unsigned int* tpack     = (unsigned int*)(ws + 395264);   // N*8*4 B

  // output layout (flat, return order): group_emb | topk_idx | distances | new_dict
  float* out      = (float*)d_out;
  float* out_ge   = out;
  float* out_ti   = out + (size_t)N * D_DIM;
  float* out_dist = out + (size_t)N * (D_DIM + TOPK);
  float* out_nd   = out + (size_t)N * (D_DIM + TOPK + C_DICT);

  prep_kernel<<<C_DICT, D_DIM, 0, stream>>>(dict, dhat, cluster, input_sum);

  const int blocks = (N + ROWS_PER_BLOCK - 1) / ROWS_PER_BLOCK;
  main_kernel<<<blocks, WAVES * 32, 0, stream>>>(
      x, dict, dhat, out_ge, out_ti, out_dist, cluster, tpack, N);

  scatter_gemm_kernel<<<NB2, 256, 0, stream>>>(x, tpack, input_sum, N);

  finalize_kernel<<<C_DICT, D_DIM, 0, stream>>>(dict, cluster, input_sum, out_nd);
}
